// CertViT_85177791414996
// MI455X (gfx1250) — compile-verified, matrix operand order checked by reading, not run on hardware
//
#include <hip/hip_runtime.h>
#include <hip/hip_bf16.h>
#include <math.h>

// ---------------- model constants ----------------
#define LAYERS  12
#define DMODEL  768
#define NHEAD   12
#define HDIM    64
#define BATCH   32
#define NP      196          // 14x14 patches
#define NTOK0   197
#define KEEP    137          // int(197*0.7)
#define NTOK1   139          // 1 + 137 + 1
#define CLSN    100
#define SEL_LAYER 3

typedef __attribute__((ext_vector_type(16))) __bf16 v16bf;
typedef __attribute__((ext_vector_type(8)))  __bf16 v8bf;   // 16 B
typedef __attribute__((ext_vector_type(4)))  __bf16 v4bf;   // 8 B
typedef __attribute__((ext_vector_type(8)))  float  v8f;

union FragBF { v16bf v; v8bf q[2]; __bf16 h[16]; };

// element e of a 16-element (8-VGPR) bf16 fragment maps to K index:
//   half 0 (lanes 0-15):  K = 0..7, 16..23 ; half 1 (lanes 16-31): K = 8..15, 24..31
__device__ __forceinline__ int kmap(int e, int hl) { return ((e < 8) ? e : e + 8) + hl * 8; }

__device__ __forceinline__ v8f wmma_bf16(const FragBF& a, const FragBF& b, v8f c) {
  return __builtin_amdgcn_wmma_f32_16x16x32_bf16(false, a.v, false, b.v, (short)0, c, false, false);
}

__device__ __forceinline__ float gelu_exact(float x) {
  return 0.5f * x * (1.0f + erff(x * 0.7071067811865475f));
}

// ---------------- generic GEMM: C = act(A[M,K] @ W[K,N] + bias) (+ resid) ----------------
// Requirements (all call sites satisfy): K % 32 == 0, N % 64 == 0. M arbitrary.
// block = 256 threads (8 waves). Block tile 256(M) x 64(N), K step 32.
// Wave w owns rows [w*32, w*32+32) (2 A-fragments) and all 4 N-subtiles:
//   per K-step: 12x ds_load_b128 feed 8 WMMAs.
// LDS rows stride 40 bf16 (80 B): 16B-aligned, 20-bank stride (conflict-free
// fragment reads). B tile stored transposed [n][k] so B fragments are
// K-contiguous like A. Staging is branch-free (row clamp) and vectorized
// (float4 global loads; A side packs to one ds_store_b64).
#define GBM 256
#define GBN 64
#define GBK 32
#define LSTR 40

__global__ void __launch_bounds__(256) gemm_bf16_kernel(
    const float* __restrict__ A, const float* __restrict__ W,
    const float* __restrict__ bias, const float* __restrict__ resid,
    float* __restrict__ C, int M, int K, int N, int act) {
  __shared__ __bf16 As[GBM][LSTR];   // [m][k]  20.0 KB
  __shared__ __bf16 Bs[GBN][LSTR];   // [n][k]   5.0 KB (transposed)

  const int t    = threadIdx.x;
  const int wave = t >> 5;
  const int lane = t & 31;
  const int hl   = lane >> 4;
  const int idx  = lane & 15;
  const int m0   = blockIdx.y * GBM;
  const int n0   = blockIdx.x * GBN;

  // staging coordinates (fixed per thread)
  const int a_r0 = t >> 3;            // 0..31, +32 per i (8 iters -> 256 rows)
  const int a_c  = (t & 7) * 4;       // 0,4,..,28
  const int b_r0 = t >> 4;            // 0..15, +16 per i (2 iters -> 32 k)
  const int b_c  = (t & 15) * 4;      // 0,4,..,60

  // precompute clamped A row pointers (row duplication is harmless)
  const float* arow[8];
#pragma unroll
  for (int i = 0; i < 8; i++) {
    int gr = m0 + a_r0 + i * 32;
    if (gr >= M) gr = M - 1;
    arow[i] = A + (size_t)gr * K + a_c;
  }
  const float* wptr = W + (size_t)b_r0 * N + n0 + b_c;

  v8f z = {0.f, 0.f, 0.f, 0.f, 0.f, 0.f, 0.f, 0.f};
  v8f acc[2][4];
#pragma unroll
  for (int s = 0; s < 2; s++)
#pragma unroll
    for (int j = 0; j < 4; j++) acc[s][j] = z;

  for (int kb = 0; kb < K; kb += GBK) {
    // stage A 256x32: one float4 load + one ds_store_b64 per row-chunk
#pragma unroll
    for (int i = 0; i < 8; i++) {
      float4 fv = *(const float4*)(arow[i] + kb);
      v4bf p;
      p[0] = (__bf16)fv.x; p[1] = (__bf16)fv.y; p[2] = (__bf16)fv.z; p[3] = (__bf16)fv.w;
      *(v4bf*)&As[a_r0 + i * 32][a_c] = p;
    }
    // stage B 32x64 -> transposed [n][k] (scalar b16 stores: LDS transpose)
#pragma unroll
    for (int i = 0; i < 2; i++) {
      float4 fv = *(const float4*)(wptr + (size_t)(kb + i * 16) * N);
      int k = b_r0 + i * 16;
      Bs[b_c + 0][k] = (__bf16)fv.x;
      Bs[b_c + 1][k] = (__bf16)fv.y;
      Bs[b_c + 2][k] = (__bf16)fv.z;
      Bs[b_c + 3][k] = (__bf16)fv.w;
    }
    __syncthreads();

    // fragment loads: two ds_load_b128 each
    FragBF a[2];
#pragma unroll
    for (int s = 0; s < 2; s++) {
      const __bf16* ap = &As[wave * 32 + s * 16 + idx][hl * 8];
      a[s].q[0] = *(const v8bf*)ap;          // K = hl*8 .. hl*8+7
      a[s].q[1] = *(const v8bf*)(ap + 16);   // K = 16+hl*8 .. 16+hl*8+7
    }
    FragBF b[4];
#pragma unroll
    for (int j = 0; j < 4; j++) {
      const __bf16* bp = &Bs[j * 16 + idx][hl * 8];
      b[j].q[0] = *(const v8bf*)bp;
      b[j].q[1] = *(const v8bf*)(bp + 16);
    }
#pragma unroll
    for (int s = 0; s < 2; s++)
#pragma unroll
      for (int j = 0; j < 4; j++) acc[s][j] = wmma_bf16(a[s], b[j], acc[s][j]);
    __syncthreads();
  }

  // store: acc[s][j] component r -> (M = w*32 + s*16 + r + 8*half, N = j*16 + idx)
#pragma unroll
  for (int s = 0; s < 2; s++) {
#pragma unroll
    for (int j = 0; j < 4; j++) {
#pragma unroll
      for (int r = 0; r < 8; r++) {
        int gr = m0 + wave * 32 + s * 16 + r + 8 * hl;
        int gc = n0 + j * 16 + idx;          // always < N (N % 64 == 0)
        if (gr < M) {
          float v = acc[s][j][r];
          if (bias)  v += bias[gc];
          if (act == 1) v = gelu_exact(v);
          if (resid) v += resid[(size_t)gr * N + gc];
          C[(size_t)gr * N + gc] = v;
        }
      }
    }
  }
}

// ---------------- patch gather: x(B,3,224,224) -> patches(B*196, 768) ----------------
__global__ void patch_gather_kernel(const float* __restrict__ x, float* __restrict__ patches) {
  int t = blockIdx.x * 256 + threadIdx.x;
  int total = BATCH * NP * 768;
  if (t >= total) return;
  int e = t % 768;
  int rp = t / 768;
  int p = rp % NP;
  int b = rp / NP;
  int c  = e >> 8;          // /256
  int pe = e & 255;
  int py = pe >> 4, px = pe & 15;
  int gy = p / 14, gx = p % 14;
  patches[t] = x[(((size_t)b * 3 + c) * 224 + (gy * 16 + py)) * 224 + (gx * 16 + px)];
}

// ---------------- assemble xt = [cls, tok] + pos ----------------
__global__ void assemble_kernel(const float* __restrict__ tok, const float* __restrict__ cls,
                                const float* __restrict__ pos, float* __restrict__ xt) {
  int t = blockIdx.x * 256 + threadIdx.x;
  int total = BATCH * NTOK0 * DMODEL;
  if (t >= total) return;
  int d = t % DMODEL;
  int rn = t / DMODEL;
  int n = rn % NTOK0;
  int b = rn / NTOK0;
  float v = (n == 0) ? cls[d] : tok[((size_t)b * NP + (n - 1)) * DMODEL + d];
  xt[t] = v + pos[(size_t)n * DMODEL + d];
}

// ---------------- layernorm: one wave per row of 768 ----------------
__global__ void ln_kernel(const float* __restrict__ x, const float* __restrict__ w,
                          const float* __restrict__ b, float* __restrict__ out, int rows) {
  int rid = blockIdx.x * 8 + (threadIdx.x >> 5);
  int lane = threadIdx.x & 31;
  if (rid >= rows) return;
  const float* xr = x + (size_t)rid * DMODEL;
  float s = 0.f;
  for (int i = lane; i < DMODEL; i += 32) s += xr[i];
#pragma unroll
  for (int m = 16; m > 0; m >>= 1) s += __shfl_xor(s, m, 32);
  float mean = s * (1.0f / DMODEL);
  float v = 0.f;
  for (int i = lane; i < DMODEL; i += 32) { float d = xr[i] - mean; v += d * d; }
#pragma unroll
  for (int m = 16; m > 0; m >>= 1) v += __shfl_xor(v, m, 32);
  float inv = rsqrtf(v * (1.0f / DMODEL) + 1e-6f);
  float* orow = out + (size_t)rid * DMODEL;
  for (int i = lane; i < DMODEL; i += 32) orow[i] = (xr[i] - mean) * inv * w[i] + b[i];
}

// ---------------- attention scores: one wave computes 16x16 tile of Q K^T * scale --------------
// Row clamp instead of zero-fill: a clamped (duplicate) Q row only corrupts an
// output row that is never stored; clamped K rows produce score columns
// k >= Ntok that softmax never reads. Loads stay contiguous -> b128 + cvt.
__global__ void attn_scores_kernel(const float* __restrict__ qkv, float* __restrict__ attn,
                                   int Ntok, int NK, float scale) {
  int bh = blockIdx.x;
  int b = bh / NHEAD, h = bh % NHEAD;
  int q0 = blockIdx.y * 16, k0 = blockIdx.z * 16;
  int lane = threadIdx.x, hl = lane >> 4, idx = lane & 15;
  const float* Qb = qkv + ((size_t)b * Ntok) * (3 * DMODEL) + h * HDIM;
  const float* Kb = Qb + DMODEL;
  int qv = q0 + idx; if (qv >= Ntok) qv = Ntok - 1;
  int kv = k0 + idx; if (kv >= Ntok) kv = Ntok - 1;
  const float* qp = Qb + (size_t)qv * (3 * DMODEL);
  const float* kp = Kb + (size_t)kv * (3 * DMODEL);
  v8f acc = {0.f, 0.f, 0.f, 0.f, 0.f, 0.f, 0.f, 0.f};
  for (int kb = 0; kb < HDIM; kb += 32) {
    FragBF a, bb;
#pragma unroll
    for (int e = 0; e < 16; e++) {
      int kk = kmap(e, hl) + kb;
      a.h[e]  = (__bf16)qp[kk];
      bb.h[e] = (__bf16)kp[kk];
    }
    acc = wmma_bf16(a, bb, acc);
  }
#pragma unroll
  for (int r = 0; r < 8; r++) {
    int q = q0 + r + 8 * hl;
    int k = k0 + idx;
    if (q < Ntok) attn[((size_t)bh * Ntok + q) * NK + k] = acc[r] * scale;
  }
}

// ---------------- softmax per row; pad [Ntok,NK) zeroed; optional strength for SEL layer ----------
__global__ void softmax_kernel(float* __restrict__ attn, float* __restrict__ strength,
                               int rows, int Ntok, int NK) {
  int rid = blockIdx.x * 8 + (threadIdx.x >> 5);
  int lane = threadIdx.x & 31;
  if (rid >= rows) return;
  float* row = attn + (size_t)rid * NK;
  float m = -1e30f;
  for (int k = lane; k < Ntok; k += 32) m = fmaxf(m, row[k]);
#pragma unroll
  for (int s = 16; s > 0; s >>= 1) m = fmaxf(m, __shfl_xor(m, s, 32));
  float sum = 0.f;
  for (int k = lane; k < Ntok; k += 32) { float e = __expf(row[k] - m); row[k] = e; sum += e; }
#pragma unroll
  for (int s = 16; s > 0; s >>= 1) sum += __shfl_xor(sum, s, 32);
  float inv = 1.0f / sum;
  float ps = 0.f;
  for (int k = lane; k < Ntok; k += 32) { float p = row[k] * inv; row[k] = p; ps += p; }
  for (int k = Ntok + lane; k < NK; k += 32) row[k] = 0.f;
  if (strength) {
#pragma unroll
    for (int s = 16; s > 0; s >>= 1) ps += __shfl_xor(ps, s, 32);
    if (lane == 0) strength[rid] = ps + (float)Ntok;   // sum(relu(attn)+1)
  }
}

// ---------------- uncertainty = mean_h (N / strength) ----------------
__global__ void uncert_kernel(const float* __restrict__ strg, float* __restrict__ u, int Ntok) {
  int t = blockIdx.x * 256 + threadIdx.x;
  if (t >= BATCH * Ntok) return;
  int b = t / Ntok, q = t % Ntok;
  float s = 0.f;
  for (int h = 0; h < NHEAD; h++) s += (float)Ntok / strg[((size_t)(b * NHEAD + h)) * Ntok + q];
  u[t] = s * (1.0f / NHEAD);
}

// ---------------- AV: one wave computes 16(q) x 64(d) of attn @ V ----------------
__global__ void attn_av_kernel(const float* __restrict__ attn, const float* __restrict__ qkv,
                               float* __restrict__ outb, int Ntok, int NK) {
  int bh = blockIdx.x;
  int b = bh / NHEAD, h = bh % NHEAD;
  int q0 = blockIdx.y * 16;
  int lane = threadIdx.x, hl = lane >> 4, idx = lane & 15;
  const float* Vb = qkv + ((size_t)b * Ntok) * (3 * DMODEL) + 2 * DMODEL + h * HDIM;
  int qv = q0 + idx; if (qv >= Ntok) qv = Ntok - 1;   // clamped row never stored
  const float* ar = attn + ((size_t)bh * Ntok + qv) * NK;
  v8f z = {0.f, 0.f, 0.f, 0.f, 0.f, 0.f, 0.f, 0.f};
  v8f acc[4]; acc[0] = z; acc[1] = z; acc[2] = z; acc[3] = z;
  for (int kb = 0; kb < NK; kb += 32) {
    FragBF a;
#pragma unroll
    for (int e = 0; e < 16; e++) a.h[e] = (__bf16)ar[kmap(e, hl) + kb];
#pragma unroll
    for (int j = 0; j < 4; j++) {
      FragBF bb;
#pragma unroll
      for (int e = 0; e < 16; e++) {
        int kk = kmap(e, hl) + kb;
        int vr = (kk < Ntok) ? kk : (Ntok - 1);   // attn pad is 0, clamp row to stay in-bounds
        bb.h[e] = (__bf16)Vb[(size_t)vr * (3 * DMODEL) + j * 16 + idx];
      }
      acc[j] = wmma_bf16(a, bb, acc[j]);
    }
  }
#pragma unroll
  for (int j = 0; j < 4; j++) {
#pragma unroll
    for (int r = 0; r < 8; r++) {
      int q = q0 + r + 8 * hl;
      if (q < Ntok)
        outb[((size_t)b * Ntok + q) * DMODEL + h * HDIM + j * 16 + idx] = acc[j][r];
    }
  }
}

// ---------------- pruning: rank-based top-K (matches top_k desc order + index tiebreak) ----------
__global__ void prune_rank_kernel(const float* __restrict__ u, int* __restrict__ sel,
                                  int* __restrict__ flag) {
  int b = blockIdx.x;
  __shared__ float su[NP];
  int t = threadIdx.x;
  if (t < NP) su[t] = u[b * NTOK0 + 1 + t];
  __syncthreads();
  if (t < NP) {
    float ui = su[t];
    int rank = 0;
    for (int j = 0; j < NP; j++) {
      float uj = su[j];
      if (uj > ui || (uj == ui && j < t)) rank++;
    }
    if (rank < KEEP) { sel[b * KEEP + rank] = t; flag[b * NP + t] = 1; }
    else flag[b * NP + t] = 0;
  }
}

__global__ void prune_gather_kernel(const float* __restrict__ xt, const int* __restrict__ sel,
                                    const int* __restrict__ flag, float* __restrict__ xt2) {
  int t = blockIdx.x * 256 + threadIdx.x;
  int total = BATCH * NTOK1 * DMODEL;
  if (t >= total) return;
  int d = t % DMODEL;
  int rp = t / DMODEL;
  int pos = rp % NTOK1;
  int b = rp / NTOK1;
  float v;
  if (pos == 0) {
    v = xt[((size_t)b * NTOK0) * DMODEL + d];
  } else if (pos <= KEEP) {
    int i = sel[b * KEEP + pos - 1];
    v = xt[((size_t)b * NTOK0 + 1 + i) * DMODEL + d];
  } else {  // stl token: sum of dropped image tokens
    float s = 0.f;
    for (int i = 0; i < NP; i++)
      if (!flag[b * NP + i]) s += xt[((size_t)b * NTOK0 + 1 + i) * DMODEL + d];
    v = s;
  }
  xt2[t] = v;
}

// ---------------- classifier head on cls token ----------------
__global__ void head_kernel(const float* __restrict__ xt2, const float* __restrict__ hw,
                            const float* __restrict__ hb, float* __restrict__ out) {
  int t = blockIdx.x * 256 + threadIdx.x;
  if (t >= BATCH * CLSN) return;
  int b = t / CLSN, c = t % CLSN;
  const float* xr = xt2 + (size_t)b * NTOK1 * DMODEL;
  float s = hb[c];
  for (int d = 0; d < DMODEL; d++) s += xr[d] * hw[(size_t)d * CLSN + c];
  out[t] = s;
}

// ---------------- host orchestration ----------------
static inline void launch_gemm(const float* A, const float* W, const float* bias,
                               const float* resid, float* C, int M, int K, int N, int act,
                               hipStream_t s) {
  dim3 grid((unsigned)((N + GBN - 1) / GBN), (unsigned)((M + GBM - 1) / GBM));
  gemm_bf16_kernel<<<grid, 256, 0, s>>>(A, W, bias, resid, C, M, K, N, act);
}

extern "C" void kernel_launch(void* const* d_in, const int* in_sizes, int n_in,
                              void* d_out, int out_size, void* d_ws, size_t ws_size,
                              hipStream_t stream) {
  const float* x       = (const float*)d_in[0];
  const float* patch_w = (const float*)d_in[1];
  const float* patch_b = (const float*)d_in[2];
  const float* cls_tok = (const float*)d_in[3];
  const float* pos_emb = (const float*)d_in[4];
  const float* ln1_w   = (const float*)d_in[5];
  const float* ln1_b   = (const float*)d_in[6];
  const float* qkv_w   = (const float*)d_in[7];
  const float* qkv_b   = (const float*)d_in[8];
  const float* proj_w  = (const float*)d_in[9];
  const float* proj_b  = (const float*)d_in[10];
  const float* ln2_w   = (const float*)d_in[11];
  const float* ln2_b   = (const float*)d_in[12];
  const float* fc1_w   = (const float*)d_in[13];
  const float* fc1_b   = (const float*)d_in[14];
  const float* fc2_w   = (const float*)d_in[15];
  const float* fc2_b   = (const float*)d_in[16];
  const float* head_w  = (const float*)d_in[17];
  const float* head_b  = (const float*)d_in[18];
  float* out = (float*)d_out;
  (void)in_sizes; (void)n_in; (void)out_size; (void)ws_size;

  // workspace carve-up (floats), ~256 MB total
  float* ws = (float*)d_ws;
  size_t off = 0;
  float* xt   = ws + off; off += (size_t)BATCH * NTOK0 * DMODEL;           // residual stream (N=197)
  float* xt2  = ws + off; off += (size_t)BATCH * NTOK1 * DMODEL;           // residual stream (N=139)
  float* hbuf = ws + off; off += (size_t)BATCH * NTOK0 * DMODEL;           // LN out / attn out
  float* qkvb = ws + off; off += (size_t)BATCH * NTOK0 * 3 * DMODEL;       // qkv (aliases patch scratch)
  float* attn = ws + off; off += (size_t)BATCH * NHEAD * NTOK0 * 224;      // attn probs (K padded)
  float* tmpb = ws + off; off += (size_t)BATCH * NTOK0 * 4 * DMODEL;       // mlp hidden / tok scratch
  float* strg = ws + off; off += (size_t)BATCH * NHEAD * NTOK0;
  float* ubuf = ws + off; off += (size_t)BATCH * NTOK0;
  int*   selb = (int*)(ws + off); off += (size_t)BATCH * KEEP;
  int*   flgb = (int*)(ws + off); off += (size_t)BATCH * NP;

  // ---- patch embed ----
  float* patches = qkvb;   // scratch alias
  float* tok     = tmpb;   // scratch alias
  {
    int total = BATCH * NP * DMODEL;
    patch_gather_kernel<<<(total + 255) / 256, 256, 0, stream>>>(x, patches);
  }
  launch_gemm(patches, patch_w, patch_b, nullptr, tok, BATCH * NP, 768, DMODEL, 0, stream);
  {
    int total = BATCH * NTOK0 * DMODEL;
    assemble_kernel<<<(total + 255) / 256, 256, 0, stream>>>(tok, cls_tok, pos_emb, xt);
  }

  float* cur = xt;
  int Ntok = NTOK0;
  for (int n = 0; n < LAYERS; n++) {
    int rows = BATCH * Ntok;
    ln_kernel<<<(rows + 7) / 8, 256, 0, stream>>>(cur, ln1_w + n * DMODEL, ln1_b + n * DMODEL,
                                                  hbuf, rows);
    launch_gemm(hbuf, qkv_w + (size_t)n * DMODEL * 3 * DMODEL, qkv_b + n * 3 * DMODEL,
                nullptr, qkvb, rows, DMODEL, 3 * DMODEL, 0, stream);

    int NK = (Ntok + 31) & ~31;
    dim3 gs((unsigned)(BATCH * NHEAD), (unsigned)((Ntok + 15) / 16), (unsigned)(NK / 16));
    attn_scores_kernel<<<gs, 32, 0, stream>>>(qkvb, attn, Ntok, NK, 0.125f);

    int srows = BATCH * NHEAD * Ntok;
    softmax_kernel<<<(srows + 7) / 8, 256, 0, stream>>>(attn, (n == SEL_LAYER) ? strg : nullptr,
                                                        srows, Ntok, NK);
    if (n == SEL_LAYER)
      uncert_kernel<<<(BATCH * Ntok + 255) / 256, 256, 0, stream>>>(strg, ubuf, Ntok);

    dim3 ga((unsigned)(BATCH * NHEAD), (unsigned)((Ntok + 15) / 16));
    attn_av_kernel<<<ga, 32, 0, stream>>>(attn, qkvb, hbuf, Ntok, NK);

    launch_gemm(hbuf, proj_w + (size_t)n * DMODEL * DMODEL, proj_b + n * DMODEL,
                cur, cur, rows, DMODEL, DMODEL, 0, stream);   // in-place residual add

    if (n == SEL_LAYER) {
      prune_rank_kernel<<<BATCH, 256, 0, stream>>>(ubuf, selb, flgb);
      int total = BATCH * NTOK1 * DMODEL;
      prune_gather_kernel<<<(total + 255) / 256, 256, 0, stream>>>(cur, selb, flgb, xt2);
      cur = xt2; Ntok = NTOK1; rows = BATCH * Ntok;
    }

    ln_kernel<<<(rows + 7) / 8, 256, 0, stream>>>(cur, ln2_w + n * DMODEL, ln2_b + n * DMODEL,
                                                  hbuf, rows);
    launch_gemm(hbuf, fc1_w + (size_t)n * DMODEL * 4 * DMODEL, fc1_b + n * 4 * DMODEL,
                nullptr, tmpb, rows, DMODEL, 4 * DMODEL, 1, stream);     // GELU
    launch_gemm(tmpb, fc2_w + (size_t)n * 4 * DMODEL * DMODEL, fc2_b + n * DMODEL,
                cur, cur, rows, 4 * DMODEL, DMODEL, 0, stream);          // residual add
  }

  head_kernel<<<(BATCH * CLSN + 255) / 256, 256, 0, stream>>>(cur, head_w, head_b, out);
}
